// ConceptDescribe_19258633356165
// MI455X (gfx1250) — compile-verified, matrix operand order checked by reading, not run on hardware
//
#include <hip/hip_runtime.h>
#include <hip/hip_bf16.h>
#include <math.h>

// ---------------- problem constants ----------------
#define CHAR_SIZE 20000
#define EMBED 128
#define HID   256
#define G4    (4 * HID)      // 1024
#define BSZ   256
#define TLEN  512
#define KT_X  4              // 128 / 32  k-tiles from x
#define KT_H  8              // 256 / 32  k-tiles from h
#define KT_ALL 12
#define NTILES 64            // 1024 / 16

typedef __attribute__((ext_vector_type(16))) __bf16 v16bf;
typedef __attribute__((ext_vector_type(8)))  float  v8f;

#define LOG2E 1.44269504088896340736f

static __device__ __forceinline__ float fast_sigmoid(float x) {
  return __builtin_amdgcn_rcpf(1.0f + __builtin_amdgcn_exp2f(-LOG2E * x));
}

static __device__ __forceinline__ float fast_tanh(float x) {
#if __has_builtin(__builtin_amdgcn_tanhf)
  return __builtin_amdgcn_tanhf(x);   // v_tanh_f32 (CDNA5 TRANS op)
#else
  return tanhf(x);
#endif
}

// ---------------- init: zero pair counters + h double buffers ----------------
__global__ void init_kernel(unsigned* counters, __bf16* h_bf) {
  int i = blockIdx.x * blockDim.x + threadIdx.x;
  if (i < 512) counters[i] = 0u;                // 32 pair counters, 64B apart
  const int HTOT = 2 * 2 * BSZ * HID;           // 2 dirs * 2 parity * B*HID
  for (int k = i; k < HTOT; k += gridDim.x * blockDim.x) h_bf[k] = (__bf16)0.0f;
}

// ---------------- embed gather + mask + fp32->bf16, time-major ----------------
__global__ void embed_mask_kernel(const int* __restrict__ inputs,
                                  const float* __restrict__ table,
                                  __bf16* __restrict__ x_bf,
                                  unsigned char* __restrict__ mask) {
  int idx = blockIdx.x * blockDim.x + threadIdx.x;   // over B*T
  if (idx >= BSZ * TLEN) return;
  int b = idx / TLEN, t = idx % TLEN;
  int tok = inputs[idx];
  mask[(size_t)t * BSZ + b] = (tok > 0) ? 1 : 0;
  const float* src = table + (size_t)tok * EMBED;
  __bf16* dst = x_bf + ((size_t)t * BSZ + b) * EMBED;
#pragma unroll 4
  for (int k = 0; k < EMBED; ++k) dst[k] = (__bf16)src[k];
}

// ---------------- pack [W;U] (384 x 1024) into WMMA B-fragment layout ----------
// layout: [dir][ktile(12)][ntile(64)][lane(32)][16 bf16]
__global__ void pack_weights_kernel(const float* __restrict__ Wf, const float* __restrict__ Uf,
                                    const float* __restrict__ Wb, const float* __restrict__ Ub,
                                    __bf16* __restrict__ wpack) {
  int lane = threadIdx.x;                  // 32
  int blk  = blockIdx.x;                   // dir*12*64 + kt*64 + nt
  int nt   = blk % NTILES;
  int kt   = (blk / NTILES) % KT_ALL;
  int dir  = blk / (NTILES * KT_ALL);
  const float* W = dir ? Wb : Wf;
  const float* U = dir ? Ub : Uf;
  int col = nt * 16 + (lane & 15);
  int kb  = kt * 32 + (lane >> 4) * 16;
  __bf16* dst = wpack + ((size_t)blk * 32 + lane) * 16;
#pragma unroll
  for (int i = 0; i < 16; ++i) {
    int k = kb + i;
    float v = (k < EMBED) ? W[(size_t)k * G4 + col]
                          : U[(size_t)(k - EMBED) * G4 + col];
    dst[i] = (__bf16)v;
  }
}

// ---------------- pairwise barrier (monotonic counter, 2 WGs) -----------------
static __device__ __forceinline__ void pairBarrier(unsigned* cnt, int step) {
  __syncthreads();
  if (threadIdx.x == 0) {
    __threadfence();
    atomicAdd(cnt, 1u);
    unsigned target = 2u * (unsigned)(step + 1);
    while (__hip_atomic_load(cnt, __ATOMIC_ACQUIRE, __HIP_MEMORY_SCOPE_AGENT) < target) {
      __builtin_amdgcn_s_sleep(1);
    }
  }
  __syncthreads();
}

#define WMMA_BF16(A, Bm, C) \
  __builtin_amdgcn_wmma_f32_16x16x32_bf16(false, (A), false, (Bm), (short)0, (C), false, false)

// ---------------- persistent bidirectional LSTM ------------------------------
// grid (32, 2): blockIdx.x = {mTile (16 batch rows), half}, blockIdx.y = dir.
// Paired WGs (half 0/1) each compute 128 hidden cols of the same batch block;
// the recurrence is batch-diagonal so only the pair must synchronize.
// Weights: 48 register-resident B-fragments per wave (8 waves/WGP fits).
__global__ __launch_bounds__(256, 1)
void lstm_persist_kernel(const __bf16* __restrict__ x_bf,
                         const unsigned char* __restrict__ mask,
                         const __bf16* __restrict__ wpack,
                         const float* __restrict__ bias_f,
                         const float* __restrict__ bias_b,
                         __bf16* __restrict__ h_bf,
                         float* __restrict__ bivalue,
                         unsigned* __restrict__ counters) {
  const int dir  = blockIdx.y;                       // 0 = fwd, 1 = bwd
  const int lane = threadIdx.x & 31;
  const int bx   = blockIdx.x;                       // 0..31
  const int mTile = bx >> 1;                         // batch tile  (16 rows)
  const int hTile = (bx & 1) * 8 + (threadIdx.x >> 5);   // hidden tile (16 cols)
  const int col   = hTile * 16 + (lane & 15);        // hidden unit this lane owns
  const int arow  = mTile * 16 + (lane & 15);        // A-fragment row for this lane
  const int ksel  = (lane >> 4) * 16;                // A-fragment K offset
  const int rowbase = mTile * 16 + (lane >> 4) * 8;  // C/D rows: j + rowbase

  const float* bias = dir ? bias_b : bias_f;
  const float bi = bias[col];
  const float bf_ = bias[HID + col];
  const float bg = bias[2 * HID + col];
  const float bo = bias[3 * HID + col];

  const __bf16* wp = wpack + (size_t)dir * (KT_ALL * NTILES * 32 * 16);
  __bf16* hdir = h_bf + (size_t)dir * (2 * BSZ * HID);
  unsigned* cnt = counters + (dir * 16 + mTile) * 16;   // 64B apart per pair

  // ---- preload all 48 weight B-fragments into registers (resident 512 steps) ----
  v16bf Bw[KT_ALL][4];
#pragma unroll
  for (int kt = 0; kt < KT_ALL; ++kt) {
    const __bf16* wb = wp + (size_t)kt * NTILES * 32 * 16 + (size_t)lane * 16;
    Bw[kt][0] = *(const v16bf*)(wb + (size_t)(hTile)      * 512);
    Bw[kt][1] = *(const v16bf*)(wb + (size_t)(hTile + 16) * 512);
    Bw[kt][2] = *(const v16bf*)(wb + (size_t)(hTile + 32) * 512);
    Bw[kt][3] = *(const v16bf*)(wb + (size_t)(hTile + 48) * 512);
  }

  float h_reg[8];
  float c_reg[8];
#pragma unroll
  for (int j = 0; j < 8; ++j) { h_reg[j] = 0.0f; c_reg[j] = 0.0f; }

  for (int step = 0; step < TLEN; ++step) {
    const int t = dir ? (TLEN - 1 - step) : step;

    // prefetch next timestep's activation row
    if (step + 1 < TLEN) {
      const int tn = dir ? (TLEN - 2 - step) : (step + 1);
      __builtin_prefetch(x_bf + ((size_t)tn * BSZ + arow) * EMBED + ksel, 0, 1);
    }

    const __bf16* xr = x_bf + ((size_t)t * BSZ + arow) * EMBED + ksel;
    const __bf16* hr = hdir + (size_t)(step & 1) * BSZ * HID + (size_t)arow * HID + ksel;

    v8f ai = {}, af = {}, ag = {}, ao = {};

    // ---- K loop: kt 0..3 from x_t, kt 4..11 from h_{t-1}; A double-buffered ----
    v16bf acur = *(const v16bf*)xr;
#pragma unroll
    for (int kt = 0; kt < KT_ALL; ++kt) {
      v16bf anext = acur;
      if (kt + 1 < KT_ALL) {
        const int kn = kt + 1;
        const __bf16* p = (kn < KT_X) ? (xr + kn * 32) : (hr + (kn - KT_X) * 32);
        anext = *(const v16bf*)p;
      }
      ai = WMMA_BF16(acur, Bw[kt][0], ai);
      af = WMMA_BF16(acur, Bw[kt][1], af);
      ag = WMMA_BF16(acur, Bw[kt][2], ag);
      ao = WMMA_BF16(acur, Bw[kt][3], ao);
      acur = anext;
    }

    // ---- gate math (hardware exp2/rcp/tanh), masked carry, publish h ----
    __bf16* hw = hdir + (size_t)((step + 1) & 1) * BSZ * HID;
    const unsigned char* mrow = mask + (size_t)t * BSZ;
#pragma unroll
    for (int j = 0; j < 8; ++j) {
      const int row = rowbase + j;
      const bool m = mrow[row] != 0;
      float zi = ai[j] + bi;
      float zf = af[j] + bf_;
      float zg = ag[j] + bg;
      float zo = ao[j] + bo;
      float cn = fast_sigmoid(zf) * c_reg[j] + fast_sigmoid(zi) * fast_tanh(zg);
      float hn = fast_sigmoid(zo) * fast_tanh(cn);
      c_reg[j] = m ? cn : c_reg[j];
      h_reg[j] = m ? hn : h_reg[j];
      hw[(size_t)row * HID + col] = (__bf16)h_reg[j];
    }

    __threadfence();
    pairBarrier(cnt, step);
  }

  // final hidden state -> bivalue[B, 2H]  (fwd cols 0..255, bwd 256..511)
#pragma unroll
  for (int j = 0; j < 8; ++j) {
    const int row = rowbase + j;
    bivalue[(size_t)row * (2 * HID) + dir * HID + col] = h_reg[j];
  }
}

// ---------------- heads: sigmoid classifier + softmax entity -----------------
__global__ void heads_kernel(const float* __restrict__ bivalue,
                             const int* __restrict__ input_end,
                             const float* __restrict__ cls_W, const float* __restrict__ cls_b,
                             const float* __restrict__ ent_W, const float* __restrict__ ent_b,
                             float* __restrict__ out) {
  int b = blockIdx.x * blockDim.x + threadIdx.x;
  if (b >= BSZ) return;

  const float* row = bivalue + (size_t)b * (2 * HID);
  float e0 = ent_b[0], e1 = ent_b[1], e2 = ent_b[2];
#pragma unroll 4
  for (int k = 0; k < 2 * HID; ++k) {
    float v = row[k];
    e0 += v * ent_W[k * 3 + 0];
    e1 += v * ent_W[k * 3 + 1];
    e2 += v * ent_W[k * 3 + 2];
  }
  float mx = fmaxf(e0, fmaxf(e1, e2));
  float x0 = __builtin_amdgcn_exp2f(LOG2E * (e0 - mx));
  float x1 = __builtin_amdgcn_exp2f(LOG2E * (e1 - mx));
  float x2 = __builtin_amdgcn_exp2f(LOG2E * (e2 - mx));
  float inv = 1.0f / (x0 + x1 + x2);
  out[BSZ + b * 3 + 0] = x0 * inv;
  out[BSZ + b * 3 + 1] = x1 * inv;
  out[BSZ + b * 3 + 2] = x2 * inv;

  const float* rowe = bivalue + (size_t)input_end[b] * (2 * HID);
  float s = cls_b[0];
#pragma unroll 4
  for (int k = 0; k < 2 * HID; ++k) s += rowe[k] * cls_W[k];
  out[b] = fast_sigmoid(s);
}

// ---------------- launch ------------------------------------------------------
extern "C" void kernel_launch(void* const* d_in, const int* in_sizes, int n_in,
                              void* d_out, int out_size, void* d_ws, size_t ws_size,
                              hipStream_t stream) {
  const int*   inputs    = (const int*)  d_in[0];
  const int*   input_end = (const int*)  d_in[1];
  const float* table     = (const float*)d_in[2];
  const float* Wf        = (const float*)d_in[3];
  const float* Uf        = (const float*)d_in[4];
  const float* bf        = (const float*)d_in[5];
  const float* Wb        = (const float*)d_in[6];
  const float* Ub        = (const float*)d_in[7];
  const float* bb        = (const float*)d_in[8];
  const float* cls_W     = (const float*)d_in[9];
  const float* cls_b     = (const float*)d_in[10];
  const float* ent_W     = (const float*)d_in[11];
  const float* ent_b     = (const float*)d_in[12];
  float* out             = (float*)d_out;

  char* ws = (char*)d_ws;
  size_t off = 0;
  auto take = [&](size_t bytes) {
    char* p = ws + off;
    off += (bytes + 255) & ~(size_t)255;
    return p;
  };
  unsigned*      counters = (unsigned*)     take(512 * sizeof(unsigned));
  __bf16*        x_bf     = (__bf16*)       take((size_t)TLEN * BSZ * EMBED * 2);
  unsigned char* mask     = (unsigned char*)take((size_t)TLEN * BSZ);
  __bf16*        wpack    = (__bf16*)       take((size_t)2 * KT_ALL * NTILES * 32 * 16 * 2);
  __bf16*        h_bf     = (__bf16*)       take((size_t)2 * 2 * BSZ * HID * 2);
  float*         bivalue  = (float*)        take((size_t)BSZ * 2 * HID * 4);
  (void)ws_size; (void)in_sizes; (void)n_in; (void)out_size;

  init_kernel<<<64, 256, 0, stream>>>(counters, h_bf);
  embed_mask_kernel<<<(BSZ * TLEN + 127) / 128, 128, 0, stream>>>(inputs, table, x_bf, mask);
  pack_weights_kernel<<<2 * KT_ALL * NTILES, 32, 0, stream>>>(Wf, Uf, Wb, Ub, wpack);
  lstm_persist_kernel<<<dim3(32, 2), 256, 0, stream>>>(
      x_bf, mask, wpack, bf, bb, h_bf, bivalue, counters);
  heads_kernel<<<1, 256, 0, stream>>>(bivalue, input_end, cls_W, cls_b, ent_W, ent_b, out);
}
